// get_model_81879256531099
// MI455X (gfx1250) — matrix operand order repaired and checked by reading, not compiled, not run on hardware
//
#include <hip/hip_runtime.h>
#include <math.h>

// ---------------------------------------------------------------------------
// PointNet++ semantic segmentation forward, CDNA5 (gfx1250, wave32, WMMA).
//
// Assumptions (cannot be runtime-verified in compile-only mode):
//  * d_in[] follows jax pytree flattening of setup_inputs() (dict keys sorted
//    alphabetically): fp1(3 layers), fp2(2), fp3(2), fp4(2),
//    head(W1,W2,b1,b2,be1,g1,m1,v1), sa1(3), sa2(3), sa3(3), sa4(3), xyz.
//    Each layer dict flattens as (W, b, be, g, m, v).
//  * ws_size >= ~56 MB.
//
// Round-2 changes:
//  * All MLP weights are pre-converted once per launch into zero-padded f16
//    (Co x Kpad, Kpad = ceil32(Cin)) so that WMMA B-fragment loads are
//    unconditional contiguous 32-byte reads (no per-element EXEC-toggled
//    guarded loads, which dominated the round-1 inner loop).
//  * Unfused GEMM A-fragment loads hand-vectorized to 4x float4.
// ---------------------------------------------------------------------------

#define BB 4
#define N0 4096
#define NS 64
#define NCLS 13
#define EPS_BN 1e-5f

typedef __attribute__((ext_vector_type(16))) _Float16 v16h;
typedef __attribute__((ext_vector_type(8)))  float    v8f;

// ---------------------------------------------------------------------------
// Weight conversion: f32 (Co x Cin) -> f16 (Co x Kpad), zero padded along K.
// ---------------------------------------------------------------------------
__global__ void k_cvt_w(const float* __restrict__ W, int Cin, int Kpad, int Co,
                        _Float16* __restrict__ Wh) {
  int gid = blockIdx.x * blockDim.x + threadIdx.x;
  if (gid >= Co * Kpad) return;
  int n = gid / Kpad;
  int k = gid - n * Kpad;
  Wh[gid] = (k < Cin) ? (_Float16)W[(size_t)n * Cin + k] : (_Float16)0.0f;
}

// ---------------------------------------------------------------------------
// transpose (B,9,N) -> pts0 (B,N,9) and xyz0 (B,N,3)
// ---------------------------------------------------------------------------
__global__ void k_transpose(const float* __restrict__ in,
                            float* __restrict__ pts0,
                            float* __restrict__ xyz0) {
  int gid = blockIdx.x * blockDim.x + threadIdx.x;
  int total = BB * 9 * N0;
  if (gid >= total) return;
  int n = gid % N0;
  int t = gid / N0;
  int c = t % 9;
  int b = t / 9;
  float v = in[gid];
  pts0[((size_t)b * N0 + n) * 9 + c] = v;
  if (c < 3) xyz0[((size_t)b * N0 + n) * 3 + c] = v;
}

// ---------------------------------------------------------------------------
// Farthest point sampling: one workgroup per batch, dist[] resident in LDS.
// ---------------------------------------------------------------------------
__global__ void __launch_bounds__(256)
k_fps(const float* __restrict__ xyz, int Nin, int npoint, int* __restrict__ fidx) {
  __shared__ float dist[N0];
  __shared__ float rmax[256];
  __shared__ int   rarg[256];
  __shared__ int   s_far;
  int b = blockIdx.x;
  const float* P = xyz + (size_t)b * Nin * 3;
  int t = threadIdx.x;
  for (int i = t; i < Nin; i += 256) dist[i] = 1e10f;
  if (t == 0) s_far = 0;
  __syncthreads();
  for (int it = 0; it < npoint; ++it) {
    int far = s_far;
    if (t == 0) fidx[b * npoint + it] = far;
    float cx = P[far * 3 + 0], cy = P[far * 3 + 1], cz = P[far * 3 + 2];
    float best = -1.0f;
    int bi = 0x7fffffff;
    for (int i = t; i < Nin; i += 256) {
      float dx = P[i * 3 + 0] - cx;
      float dy = P[i * 3 + 1] - cy;
      float dz = P[i * 3 + 2] - cz;
      float d = dx * dx + dy * dy + dz * dz;
      float nd = fminf(dist[i], d);
      dist[i] = nd;
      if (nd > best) { best = nd; bi = i; }
    }
    rmax[t] = best;
    rarg[t] = bi;
    __syncthreads();
    for (int s = 128; s > 0; s >>= 1) {
      if (t < s) {
        if (rmax[t + s] > rmax[t] ||
            (rmax[t + s] == rmax[t] && rarg[t + s] < rarg[t])) {
          rmax[t] = rmax[t + s];
          rarg[t] = rarg[t + s];
        }
      }
      __syncthreads();
    }
    if (t == 0) s_far = rarg[0];
    __syncthreads();
  }
}

// gather centroids (3 floats each)
__global__ void k_gather3(const float* __restrict__ xyz, const int* __restrict__ fidx,
                          int Nin, int npoint, float* __restrict__ out) {
  int gid = blockIdx.x * blockDim.x + threadIdx.x;
  if (gid >= BB * npoint) return;
  int b = gid / npoint;
  int src = fidx[gid];
  const float* s = xyz + ((size_t)b * Nin + src) * 3;
  float* d = out + (size_t)gid * 3;
  d[0] = s[0]; d[1] = s[1]; d[2] = s[2];
}

// ---------------------------------------------------------------------------
// Ball query: scan in ascending index order == reference sort semantics.
// ---------------------------------------------------------------------------
__global__ void k_ball_query(const float* __restrict__ xyz,
                             const float* __restrict__ cent,
                             int Nin, int npoint, float r2,
                             int* __restrict__ gidx) {
  int gid = blockIdx.x * blockDim.x + threadIdx.x;
  if (gid >= BB * npoint) return;
  int b = gid / npoint;
  const float* P = xyz + (size_t)b * Nin * 3;
  const float* c = cent + (size_t)gid * 3;
  float cx = c[0], cy = c[1], cz = c[2];
  int* out = gidx + (size_t)gid * NS;
  int cnt = 0, first = 0;
  bool have = false;
  for (int i = 0; i < Nin; ++i) {
    float dx = P[i * 3 + 0] - cx;
    float dy = P[i * 3 + 1] - cy;
    float dz = P[i * 3 + 2] - cz;
    float d = dx * dx + dy * dy + dz * dz;
    if (d <= r2) {
      if (!have) { first = i; have = true; }
      if (cnt < NS) out[cnt++] = i;
      else break;
    }
  }
  for (; cnt < NS; ++cnt) out[cnt] = first;
}

// ---------------------------------------------------------------------------
// Fused set-abstraction MLP: one workgroup (128 threads / 4 waves) per
// centroid. Grouped 64 x Cin tile lives in LDS as f16; up to 3 layers of
// WMMA GEMM (f16 in, f32 accum) with fused BN+ReLU, ping-ponging two LDS
// buffers; max-pool over the 64 samples at the end. Grouped tensors never
// touch HBM. Weights arrive pre-converted/padded to f16 (Co x K, K%32==0),
// so B-fragment loads are branch-free contiguous 32B reads.
// ---------------------------------------------------------------------------
struct SAParams {
  const _Float16* W[3];   // padded f16 weights, row stride = ceil32(cin)
  const float* bias[3];
  const float* gamma[3];
  const float* beta[3];
  const float* mean[3];
  const float* var[3];
  int cin[3];
  int cout[3];
  int nlayers;
};

__global__ void __launch_bounds__(128)
k_sa_fused(const float* __restrict__ xyz, const float* __restrict__ feats, int featC,
           const float* __restrict__ cent, const int* __restrict__ gidx,
           int Nin, int npoint, SAParams sp, float* __restrict__ pooled, int maxdim) {
  extern __shared__ char smem_raw[];
  _Float16* bufA = (_Float16*)smem_raw;
  _Float16* bufB = bufA + 64 * maxdim;

  int blk = blockIdx.x;
  int b = blk / npoint;
  const int* gi = gidx + (size_t)blk * NS;
  const float* c = cent + (size_t)blk * 3;
  float cx = c[0], cy = c[1], cz = c[2];
  int tid = threadIdx.x;

  // Build grouped input tile (64 x K0pad) in LDS, zero-padded along K.
  int cin0 = sp.cin[0];
  int K0 = (cin0 + 31) & ~31;
  for (int e = tid; e < 64 * K0; e += 128) {
    int s = e / K0;
    int k = e - s * K0;
    float val = 0.0f;
    if (k < cin0) {
      int src = gi[s];
      if (k < 3) {
        float xv = xyz[((size_t)b * Nin + src) * 3 + k];
        val = xv - (k == 0 ? cx : (k == 1 ? cy : cz));
      } else {
        val = feats[((size_t)b * Nin + src) * featC + (k - 3)];
      }
    }
    bufA[s * K0 + k] = (_Float16)val;
  }
  __syncthreads();

  int wave = tid >> 5, lane = tid & 31;
  int r16 = lane & 15, half = lane >> 4;

  for (int l = 0; l < sp.nlayers; ++l) {
    int K = (sp.cin[l] + 31) & ~31;
    int Co = sp.cout[l];
    const _Float16* X = (l & 1) ? bufB : bufA;
    _Float16* Y = (l & 1) ? bufA : bufB;
    const _Float16* Wl = sp.W[l];
    // Each wave owns full 64-row column strips: 4 m-tiles x 1 n-tile.
    for (int nt = wave; nt < (Co >> 4); nt += 4) {
      int n = nt * 16 + r16;
      const _Float16* wrow = Wl + (size_t)n * K;
      v8f acc[4] = {};
      for (int k0 = 0; k0 < K; k0 += 32) {
        // B fragment: lane = column n, element e -> K = k0 + 16*half + e.
        v16h bf;
        const _Float16* wp = wrow + k0 + 16 * half;
#pragma unroll
        for (int e = 0; e < 16; ++e) bf[e] = wp[e];
#pragma unroll
        for (int mt = 0; mt < 4; ++mt) {
          // A fragment: lane = row, element e -> K = k0 + 16*half + e.
          v16h af;
          const _Float16* xr = X + (mt * 16 + r16) * K;
          int base = k0 + 16 * half;
#pragma unroll
          for (int e = 0; e < 8; ++e) af[e] = xr[base + e];
#pragma unroll
          for (int e = 0; e < 8; ++e) af[8 + e] = xr[base + 8 + e];
          acc[mt] = __builtin_amdgcn_wmma_f32_16x16x32_f16(
              false, af, false, bf, (short)0, acc[mt], false, false);
        }
      }
      // Fused BN + ReLU epilogue, write f16 activation tile.
      float sc = sp.gamma[l][n] * rsqrtf(sp.var[l][n] + EPS_BN);
      float sh = (sp.bias[l][n] - sp.mean[l][n]) * sc + sp.beta[l][n];
#pragma unroll
      for (int mt = 0; mt < 4; ++mt)
#pragma unroll
        for (int v = 0; v < 8; ++v) {
          float y = fmaxf(acc[mt][v] * sc + sh, 0.0f);
          Y[(mt * 16 + v + 8 * half) * Co + n] = (_Float16)y;
        }
    }
    __syncthreads();
  }

  // Max-pool over the 64 samples.
  int lf = sp.nlayers - 1;
  const _Float16* Yf = (lf & 1) ? bufA : bufB;
  int Cf = sp.cout[lf];
  for (int ch = tid; ch < Cf; ch += 128) {
    float mx = -1e30f;
    for (int s = 0; s < 64; ++s) mx = fmaxf(mx, (float)Yf[s * Cf + ch]);
    pooled[(size_t)blk * Cf + ch] = mx;
  }
}

// ---------------------------------------------------------------------------
// Unfused WMMA GEMM + BN + ReLU for FP MLPs / head layer 1.
// Requires M % 16 == 0, K % 32 == 0, Co % 16 == 0 (true for all call sites).
// W is pre-converted f16 (Co x K). One wave per 16x16 tile; 8 waves/block.
// ---------------------------------------------------------------------------
__global__ void __launch_bounds__(256)
k_gemm_bnrelu(const float* __restrict__ A, int M, int K,
              const _Float16* __restrict__ W,
              const float* __restrict__ bias,
              const float* __restrict__ gamma,
              const float* __restrict__ beta,
              const float* __restrict__ mean,
              const float* __restrict__ var,
              float* __restrict__ C, int Co) {
  int wave = threadIdx.x >> 5, lane = threadIdx.x & 31;
  int mt = blockIdx.x * 8 + wave;
  if (mt * 16 >= M) return;  // wave-uniform exit
  int nt = blockIdx.y;
  int r16 = lane & 15, half = lane >> 4;
  int m = mt * 16 + r16;
  int n = nt * 16 + r16;
  const float* arow = A + (size_t)m * K;
  const _Float16* wrow = W + (size_t)n * K;
  v8f acc = {};
  for (int k0 = 0; k0 < K; k0 += 32) {
    v16h af, bf;
    // A fragment: elements 0..7 <- arow[base..base+8),
    //             elements 8..15 <- arow[base+8..base+16)
    int base = k0 + 16 * half;
    float4 ta = *(const float4*)(arow + base);
    float4 tb = *(const float4*)(arow + base + 4);
    float4 tc = *(const float4*)(arow + base + 8);
    float4 td = *(const float4*)(arow + base + 12);
    af[0] = (_Float16)ta.x;  af[1] = (_Float16)ta.y;
    af[2] = (_Float16)ta.z;  af[3] = (_Float16)ta.w;
    af[4] = (_Float16)tb.x;  af[5] = (_Float16)tb.y;
    af[6] = (_Float16)tb.z;  af[7] = (_Float16)tb.w;
    af[8]  = (_Float16)tc.x; af[9]  = (_Float16)tc.y;
    af[10] = (_Float16)tc.z; af[11] = (_Float16)tc.w;
    af[12] = (_Float16)td.x; af[13] = (_Float16)td.y;
    af[14] = (_Float16)td.z; af[15] = (_Float16)td.w;
    const _Float16* wp = wrow + k0 + 16 * half;
#pragma unroll
    for (int e = 0; e < 16; ++e) bf[e] = wp[e];
    acc = __builtin_amdgcn_wmma_f32_16x16x32_f16(
        false, af, false, bf, (short)0, acc, false, false);
  }
  float sc = gamma[n] * rsqrtf(var[n] + EPS_BN);
  float sh = (bias[n] - mean[n]) * sc + beta[n];
#pragma unroll
  for (int v = 0; v < 8; ++v) {
    int mm = mt * 16 + v + 8 * half;
    C[(size_t)mm * Co + n] = fmaxf(acc[v] * sc + sh, 0.0f);
  }
}

// ---------------------------------------------------------------------------
// 3-NN + inverse-distance weights (feature propagation).
// ---------------------------------------------------------------------------
__global__ void k_knn3(const float* __restrict__ xyz1, int N1,
                       const float* __restrict__ xyz2, int N2,
                       int* __restrict__ kidx, float* __restrict__ kw, int M) {
  int gid = blockIdx.x * blockDim.x + threadIdx.x;
  if (gid >= M) return;
  int b = gid / N1;
  const float* p = xyz1 + (size_t)gid * 3;
  float px = p[0], py = p[1], pz = p[2];
  const float* Q = xyz2 + (size_t)b * N2 * 3;
  float d0 = 3e38f, d1 = 3e38f, d2 = 3e38f;
  int i0 = 0, i1 = 0, i2 = 0;
  for (int j = 0; j < N2; ++j) {
    float dx = Q[j * 3 + 0] - px;
    float dy = Q[j * 3 + 1] - py;
    float dz = Q[j * 3 + 2] - pz;
    float d = dx * dx + dy * dy + dz * dz;
    if (d < d0)      { d2 = d1; i2 = i1; d1 = d0; i1 = i0; d0 = d; i0 = j; }
    else if (d < d1) { d2 = d1; i2 = i1; d1 = d;  i1 = j; }
    else if (d < d2) { d2 = d;  i2 = j; }
  }
  float w0 = 1.0f / (d0 + 1e-8f);
  float w1 = 1.0f / (d1 + 1e-8f);
  float w2 = 1.0f / (d2 + 1e-8f);
  float s = w0 + w1 + w2;
  kidx[gid * 3 + 0] = i0; kidx[gid * 3 + 1] = i1; kidx[gid * 3 + 2] = i2;
  kw[gid * 3 + 0] = w0 / s; kw[gid * 3 + 1] = w1 / s; kw[gid * 3 + 2] = w2 / s;
}

// Build FP MLP input: [points1 | interpolated points2]
__global__ void k_fp_build(const float* __restrict__ pts1, int C1,
                           const float* __restrict__ pts2, int C2,
                           int N1, int N2,
                           const int* __restrict__ kidx, const float* __restrict__ kw,
                           float* __restrict__ X, int M) {
  size_t gid = (size_t)blockIdx.x * blockDim.x + threadIdx.x;
  int Ct = C1 + C2;
  if (gid >= (size_t)M * Ct) return;
  int row = (int)(gid / Ct);
  int cch = (int)(gid - (size_t)row * Ct);
  float val;
  if (cch < C1) {
    val = pts1[(size_t)row * C1 + cch];
  } else {
    int b = row / N1;
    int cc = cch - C1;
    const int* id = kidx + (size_t)row * 3;
    const float* w = kw + (size_t)row * 3;
    const float* P2 = pts2 + (size_t)b * N2 * C2;
    val = w[0] * P2[(size_t)id[0] * C2 + cc] +
          w[1] * P2[(size_t)id[1] * C2 + cc] +
          w[2] * P2[(size_t)id[2] * C2 + cc];
  }
  X[gid] = val;
}

// Head: 13 logits + log_softmax.
__global__ void k_head_logits(const float* __restrict__ X, const float* __restrict__ W2,
                              const float* __restrict__ b2, float* __restrict__ y, int M) {
  int gid = blockIdx.x * blockDim.x + threadIdx.x;
  if (gid >= M) return;
  const float* x = X + (size_t)gid * 128;
  float lg[NCLS];
  float mx = -3e38f;
  for (int o = 0; o < NCLS; ++o) {
    float s = b2[o];
    const float* w = W2 + o * 128;
    for (int cc = 0; cc < 128; ++cc) s += x[cc] * w[cc];
    lg[o] = s;
    mx = fmaxf(mx, s);
  }
  float se = 0.0f;
  for (int o = 0; o < NCLS; ++o) se += __expf(lg[o] - mx);
  float lse = mx + __logf(se);
  for (int o = 0; o < NCLS; ++o) y[(size_t)gid * NCLS + o] = lg[o] - lse;
}

// p4 (B,32,1024) -> out (B,1024,32)
__global__ void k_p4_out(const float* __restrict__ p4, float* __restrict__ out) {
  int gid = blockIdx.x * blockDim.x + threadIdx.x;
  int total = BB * 32 * 1024;
  if (gid >= total) return;
  int p = gid % 32;
  int t = gid / 32;
  int c = t % 1024;
  int b = t / 1024;
  out[gid] = p4[((size_t)b * 32 + p) * 1024 + c];
}

// ---------------------------------------------------------------------------
// Host orchestration
// ---------------------------------------------------------------------------
struct LayerPtrs { const float *W, *b, *be, *g, *m, *v; };

extern "C" void kernel_launch(void* const* d_in, const int* in_sizes, int n_in,
                              void* d_out, int out_size, void* d_ws, size_t ws_size,
                              hipStream_t stream) {
  (void)in_sizes; (void)n_in; (void)out_size; (void)ws_size;

  // ---- map inputs (jax pytree alphabetical order) ----
  int ip = 0;
  auto F = [&](void) -> const float* { return (const float*)d_in[ip++]; };
  LayerPtrs fp1[3], fp2[2], fp3[2], fp4[2], sa1[3], sa2[3], sa3[3], sa4[3];
  auto grab = [&](LayerPtrs* L, int nl) {
    for (int i = 0; i < nl; ++i) {
      L[i].W = F(); L[i].b = F(); L[i].be = F(); L[i].g = F(); L[i].m = F(); L[i].v = F();
    }
  };
  grab(fp1, 3); grab(fp2, 2); grab(fp3, 2); grab(fp4, 2);
  const float* hW1 = F(); const float* hW2 = F();
  const float* hb1 = F(); const float* hb2 = F();
  const float* hbe1 = F(); const float* hg1 = F();
  const float* hm1 = F(); const float* hv1 = F();
  grab(sa1, 3); grab(sa2, 3); grab(sa3, 3); grab(sa4, 3);
  const float* xyz_in = F();

  // ---- workspace bump allocator ----
  char* wsb = (char*)d_ws;
  size_t off = 0;
  auto alloc = [&](size_t bytes) -> void* {
    void* p = wsb + off;
    off = (off + bytes + 255) & ~(size_t)255;
    return p;
  };
  float* pts0 = (float*)alloc((size_t)BB * N0 * 9 * 4);
  float* xyz0 = (float*)alloc((size_t)BB * N0 * 3 * 4);
  float* x1 = (float*)alloc((size_t)BB * 2048 * 3 * 4);
  float* x2 = (float*)alloc((size_t)BB * 512 * 3 * 4);
  float* x3 = (float*)alloc((size_t)BB * 128 * 3 * 4);
  float* x4 = (float*)alloc((size_t)BB * 32 * 3 * 4);
  float* p1 = (float*)alloc((size_t)BB * 2048 * 128 * 4);
  float* p2 = (float*)alloc((size_t)BB * 512 * 256 * 4);
  float* p3 = (float*)alloc((size_t)BB * 128 * 512 * 4);
  float* p4 = (float*)alloc((size_t)BB * 32 * 1024 * 4);
  float* p3n = (float*)alloc((size_t)BB * 128 * 512 * 4);
  float* p2n = (float*)alloc((size_t)BB * 512 * 256 * 4);
  float* p1n = (float*)alloc((size_t)BB * 2048 * 128 * 4);
  int* fidx = (int*)alloc((size_t)BB * 2048 * 4);
  int* gidx = (int*)alloc((size_t)BB * 2048 * NS * 4);
  int* kidx = (int*)alloc((size_t)BB * N0 * 3 * 4);
  float* kw = (float*)alloc((size_t)BB * N0 * 3 * 4);
  float* fpX = (float*)alloc((size_t)8192 * 384 * 4);     // max FP input (M*Ct)
  float* midA = (float*)alloc((size_t)16384 * 128 * 4);   // max intermediate
  float* midB = (float*)alloc((size_t)16384 * 128 * 4);

  // ---- weight pre-conversion to padded f16 ----
  auto cvt = [&](const float* W, int Cin, int Co) -> const _Float16* {
    int Kpad = (Cin + 31) & ~31;
    _Float16* Wh = (_Float16*)alloc((size_t)Co * Kpad * 2);
    int tot = Co * Kpad;
    k_cvt_w<<<(tot + 255) / 256, 256, 0, stream>>>(W, Cin, Kpad, Co, Wh);
    return Wh;
  };

  // ---- stage 0: transpose ----
  {
    int tot = BB * 9 * N0;
    k_transpose<<<(tot + 255) / 256, 256, 0, stream>>>(xyz_in, pts0, xyz0);
  }

  // ---- SA stages (fused) ----
  auto run_sa = [&](const float* xyzL, int Nin, const float* feats, int featC,
                    int npoint, float radius, LayerPtrs* Ls, int nl, const int* couts,
                    float* x_out, float* pooled) {
    k_fps<<<BB, 256, 0, stream>>>(xyzL, Nin, npoint, fidx);
    int tot = BB * npoint;
    k_gather3<<<(tot + 255) / 256, 256, 0, stream>>>(xyzL, fidx, Nin, npoint, x_out);
    k_ball_query<<<(tot + 127) / 128, 128, 0, stream>>>(xyzL, x_out, Nin, npoint,
                                                        radius * radius, gidx);
    SAParams sp{};
    int cin = 3 + featC;
    int maxdim = 0;
    for (int i = 0; i < nl; ++i) {
      sp.W[i] = cvt(Ls[i].W, cin, couts[i]);
      sp.bias[i] = Ls[i].b; sp.gamma[i] = Ls[i].g;
      sp.beta[i] = Ls[i].be; sp.mean[i] = Ls[i].m; sp.var[i] = Ls[i].v;
      sp.cin[i] = cin; sp.cout[i] = couts[i];
      int kp = (cin + 31) & ~31;
      if (kp > maxdim) maxdim = kp;
      if (couts[i] > maxdim) maxdim = couts[i];
      cin = couts[i];
    }
    sp.nlayers = nl;
    size_t shmem = (size_t)2 * 64 * maxdim * 2;  // two f16 ping-pong buffers
    k_sa_fused<<<BB * npoint, 128, shmem, stream>>>(xyzL, feats, featC, x_out, gidx,
                                                    Nin, npoint, sp, pooled, maxdim);
  };

  int c1[3] = {32, 64, 128};
  int c2[3] = {128, 128, 256};
  int c3[3] = {256, 256, 512};
  int c4[3] = {512, 512, 1024};
  run_sa(xyz0, 4096, pts0, 9,   2048, 0.1f, sa1, 3, c1, x1, p1);
  run_sa(x1,   2048, p1,   128, 512,  0.2f, sa2, 3, c2, x2, p2);
  run_sa(x2,   512,  p2,   256, 128,  0.4f, sa3, 3, c3, x3, p3);
  run_sa(x3,   128,  p3,   512, 32,   0.8f, sa4, 3, c4, x4, p4);

  // ---- FP stages ----
  auto run_fp = [&](const float* xyzA, int N1, const float* xyzBp, int N2,
                    const float* pts1, int C1, const float* pts2, int C2,
                    LayerPtrs* Ls, int nl, const int* couts, float* outbuf) {
    int M = BB * N1;
    k_knn3<<<(M + 127) / 128, 128, 0, stream>>>(xyzA, N1, xyzBp, N2, kidx, kw, M);
    int Ct = C1 + C2;
    size_t tot = (size_t)M * Ct;
    k_fp_build<<<(unsigned)((tot + 255) / 256), 256, 0, stream>>>(
        pts1, C1, pts2, C2, N1, N2, kidx, kw, fpX, M);
    const float* in = fpX;
    int K = Ct;
    for (int l = 0; l < nl; ++l) {
      float* out = (l == nl - 1) ? outbuf : ((l & 1) ? midB : midA);
      int Co = couts[l];
      const _Float16* Wh = cvt(Ls[l].W, K, Co);  // K%32==0, no padding
      dim3 g((M / 16 + 7) / 8, Co / 16);
      k_gemm_bnrelu<<<g, 256, 0, stream>>>(in, M, K, Wh, Ls[l].b, Ls[l].g,
                                           Ls[l].be, Ls[l].m, Ls[l].v, out, Co);
      in = out;
      K = Co;
    }
  };

  int f4[2] = {512, 512};
  int f3[2] = {512, 256};
  int f2[2] = {256, 128};
  int f1[3] = {128, 128, 128};
  run_fp(x3, 128, x4, 32,  p3, 512, p4, 1024, fp4, 2, f4, p3n);
  run_fp(x2, 512, x3, 128, p2, 256, p3n, 512, fp3, 2, f3, p2n);
  run_fp(x1, 2048, x2, 512, p1, 128, p2n, 256, fp2, 2, f2, p1n);
  // fp1: 3 layers -> fpX->midA->midB->midA (p0 lives in midA)
  run_fp(xyz0, 4096, x1, 2048, nullptr, 0, p1n, 128, fp1, 3, f1, midA);

  // ---- head ----
  {
    int M = BB * N0;  // 16384
    const _Float16* W1h = cvt(hW1, 128, 128);
    dim3 g((M / 16 + 7) / 8, 128 / 16);
    k_gemm_bnrelu<<<g, 256, 0, stream>>>(midA, M, 128, W1h, hb1, hg1, hbe1, hm1, hv1,
                                         midB, 128);
    k_head_logits<<<(M + 127) / 128, 128, 0, stream>>>(midB, hW2, hb2, (float*)d_out, M);
    int tot = BB * 32 * 1024;
    k_p4_out<<<(tot + 255) / 256, 256, 0, stream>>>(p4,
                                                    (float*)d_out + (size_t)BB * N0 * NCLS);
  }
}
